// AlgebraMultiheadAttention_46712064312044
// MI455X (gfx1250) — matrix operand order repaired and checked by reading, not compiled, not run on hardware
//
#include <hip/hip_runtime.h>
#include <hip/hip_bf16.h>

// ---------------------------------------------------------------------------
// AlgebraMultiheadAttention on gfx1250 (MI455X), f16 WMMA + fp32 accumulate.
//
// Pipeline (all intermediates in d_ws, ~48.5 MiB, L2-resident):
//   1. x fp32 -> Xh f16 row-major [4096][1024]
//   2. mask int32 [2048][2048] -> bitmask [2048][64] u32
//   3. Wq/Wk/Wv/Wo [256,256,4] -> dense quaternion matrices M f16 [1024][1024]
//   4. Q = (Mq x + bq)*scale*log2e, K = Mk x + bk   (transposed GEMM, D[m=out,n=tok])
//      V^T = (Mv x + bv)^T                          (normal GEMM, D[m=tok,n=out])
//   5. flash attention per (b,h,qtile): scores^T = K Q^T, ctx^T = V^T P^T
//      (K/V panels staged into LDS via GLOBAL_LOAD_ASYNC_TO_LDS when available)
//   6. out = Mo ctx + bo (fp32)
// ---------------------------------------------------------------------------

typedef __attribute__((ext_vector_type(16))) _Float16     v16h;
typedef __attribute__((ext_vector_type(8)))  float        v8f;
typedef __attribute__((ext_vector_type(8)))  unsigned int v8u;
typedef __attribute__((ext_vector_type(4)))  int          v4i;

// Async global->LDS DMA path (gfx1250): guarded so the file compiles on any
// toolchain; falls back to VGPR-bounce staging otherwise.
#if __has_builtin(__builtin_amdgcn_global_load_async_to_lds_b128) && \
    __has_builtin(__builtin_amdgcn_s_wait_asynccnt)
#define USE_ASYNC_LDS 1
typedef __attribute__((address_space(1))) v4i* as1_v4i_p;  // "__device__" in diags
typedef __attribute__((address_space(3))) v4i* as3_v4i_p;  // "__shared__" in diags
#else
#define USE_ASYNC_LDS 0
#endif

__device__ __forceinline__ unsigned int pkh(float lo, float hi) {
  union { _Float16 h[2]; unsigned int u; } t;
  t.h[0] = (_Float16)lo; t.h[1] = (_Float16)hi;
  return t.u;
}

// A operand, 16x32 f16: lane m = L&15, half = L>>4.
// VGPR v holds K pair at k0 = 16*half + 2*v.
// (one contiguous 32B run)
__device__ __forceinline__ v16h load_a(const _Float16* __restrict__ base,
                                       size_t row, int ld, int colbase, int half) {
  const _Float16* r = base + row * (size_t)ld + colbase;
  v8u u;
#pragma unroll
  for (int v = 0; v < 8; ++v) {
    int k0 = half * 16 + (v << 1);
    u[v] = *reinterpret_cast<const unsigned int*>(r + k0);
  }
  return __builtin_bit_cast(v16h, u);
}

// B operand, 32x16 f16: lane n = L&15; lanes 0-15 hold K=0..15, lanes 16-31 K=16..31.
// VGPR v holds K pair at k0 = 16*half + 2*v (one contiguous 32B run).
__device__ __forceinline__ v16h load_b(const _Float16* __restrict__ base,
                                       size_t row, int ld, int colbase, int half) {
  const _Float16* r = base + row * (size_t)ld + colbase + half * 16;
  v8u u;
#pragma unroll
  for (int v = 0; v < 8; ++v)
    u[v] = *reinterpret_cast<const unsigned int*>(r + 2 * v);
  return __builtin_bit_cast(v16h, u);
}

__device__ __forceinline__ v8f wmma_f16(v16h a, v16h b, v8f c) {
  return __builtin_amdgcn_wmma_f32_16x16x32_f16(false, a, false, b, (short)0, c,
                                                false, false);
}

// ---------------------------------------------------------------------------
// 1) x fp32 -> f16 row-major.  1M threads, 4 elems each.
__global__ void k_prep_x(const float* __restrict__ x, _Float16* __restrict__ xh) {
  size_t i = (size_t)blockIdx.x * blockDim.x + threadIdx.x;
  const float4 f = reinterpret_cast<const float4*>(x)[i];
  uint2 o; o.x = pkh(f.x, f.y); o.y = pkh(f.z, f.w);
  reinterpret_cast<uint2*>(xh)[i] = o;
}

// 2) mask -> bitmask words. word w = q*64 + c covers keys c*32..c*32+31 of row q.
__global__ void k_maskbits(const int* __restrict__ mask, unsigned int* __restrict__ mb) {
  int w = blockIdx.x * blockDim.x + threadIdx.x;   // 0..131071
  const int* p = mask + (size_t)w * 32;
  unsigned int bits = 0;
#pragma unroll
  for (int j = 0; j < 32; ++j) bits |= (p[j] != 0 ? 1u : 0u) << j;
  mb[w] = bits;
}

// 3) quaternion expansion: M[(o*4+k)][ (n*4+j) ] = sgn[k][j] * W[o][n][idx[k][j]]
__global__ void k_expand(const float* __restrict__ W, _Float16* __restrict__ M) {
  int t = blockIdx.x * blockDim.x + threadIdx.x;   // 0..65535 = (o,n)
  int o = t >> 8, n = t & 255;
  const float4 w4 = reinterpret_cast<const float4*>(W)[t];
  float a[4] = {w4.x, w4.y, w4.z, w4.w};
  const int   idx[4][4] = {{0,1,2,3},{1,0,3,2},{2,3,0,1},{3,2,1,0}};
  const float sgn[4][4] = {{1.f,-1.f,-1.f,-1.f},{1.f,1.f,-1.f,1.f},
                           {1.f,1.f,1.f,-1.f},{1.f,-1.f,1.f,1.f}};
#pragma unroll
  for (int k = 0; k < 4; ++k) {
    uint2 st;
    st.x = pkh(sgn[k][0]*a[idx[k][0]], sgn[k][1]*a[idx[k][1]]);
    st.y = pkh(sgn[k][2]*a[idx[k][2]], sgn[k][3]*a[idx[k][3]]);
    *reinterpret_cast<uint2*>(M + (size_t)(o*4 + k)*1024 + n*4) = st;
  }
}

// ---------------------------------------------------------------------------
// 4a) Transposed GEMM: D[m=out,n=tok] = sum_in M[out,in]*Act[tok,in]  (+bias)*scale
//     Wave computes a 32x32 block (2x2 WMMA tiles).  Store row-major [tok][1024].
template <bool F32OUT>
__global__ __launch_bounds__(128) void k_gemm_T(const _Float16* __restrict__ Wm,
                                                const _Float16* __restrict__ Act,
                                                const float* __restrict__ bias,
                                                float postScale,
                                                void* __restrict__ outPtr) {
  int wave = blockIdx.x * 4 + (threadIdx.x >> 5);
  int lane = threadIdx.x & 31;
  int m = lane & 15, half = lane >> 4;
  int outB = wave & 31;    // 32 blocks of 32 outputs
  int tokB = wave >> 5;    // 128 blocks of 32 tokens
  v8f acc[2][2] = {};
  for (int kb = 0; kb < 32; ++kb) {
    int cb = kb * 32;
    v16h A0 = load_a(Wm,  (size_t)(outB*32 +      m), 1024, cb, half);
    v16h A1 = load_a(Wm,  (size_t)(outB*32 + 16 + m), 1024, cb, half);
    v16h B0 = load_b(Act, (size_t)(tokB*32 +      m), 1024, cb, half);
    v16h B1 = load_b(Act, (size_t)(tokB*32 + 16 + m), 1024, cb, half);
    acc[0][0] = wmma_f16(A0, B0, acc[0][0]);
    acc[0][1] = wmma_f16(A0, B1, acc[0][1]);
    acc[1][0] = wmma_f16(A1, B0, acc[1][0]);
    acc[1][1] = wmma_f16(A1, B1, acc[1][1]);
  }
#pragma unroll
  for (int i = 0; i < 2; ++i) {
    int ob = outB*32 + i*16 + half*8;
    float4 bv0 = *reinterpret_cast<const float4*>(bias + ob);
    float4 bv1 = *reinterpret_cast<const float4*>(bias + ob + 4);
    float bb[8] = {bv0.x,bv0.y,bv0.z,bv0.w,bv1.x,bv1.y,bv1.z,bv1.w};
#pragma unroll
    for (int j = 0; j < 2; ++j) {
      size_t token = (size_t)(tokB*32 + j*16 + m);
      float r[8];
#pragma unroll
      for (int e = 0; e < 8; ++e) r[e] = (acc[i][j][e] + bb[e]) * postScale;
      if constexpr (F32OUT) {
        float* out = (float*)outPtr;
        float4 f0 = {r[0],r[1],r[2],r[3]}, f1 = {r[4],r[5],r[6],r[7]};
        *reinterpret_cast<float4*>(out + token*1024 + ob)     = f0;
        *reinterpret_cast<float4*>(out + token*1024 + ob + 4) = f1;
      } else {
        _Float16* out = (_Float16*)outPtr;
        uint4 st = {pkh(r[0],r[1]), pkh(r[2],r[3]), pkh(r[4],r[5]), pkh(r[6],r[7])};
        *reinterpret_cast<uint4*>(out + token*1024 + ob) = st;
      }
    }
  }
}

// 4b) Normal GEMM for V: D[m=tok,n=out] = sum_in Act[tok,in]*Mv[out,in] + bias[out]
//     Stored transposed: VT[out][tok], f16.
__global__ __launch_bounds__(128) void k_gemm_N(const _Float16* __restrict__ Act,
                                                const _Float16* __restrict__ Wm,
                                                const float* __restrict__ bias,
                                                _Float16* __restrict__ OutT) {
  int wave = blockIdx.x * 4 + (threadIdx.x >> 5);
  int lane = threadIdx.x & 31;
  int m = lane & 15, half = lane >> 4;
  int outB = wave & 31;
  int tokB = wave >> 5;
  v8f acc[2][2] = {};  // [tok tile][out tile]
  for (int kb = 0; kb < 32; ++kb) {
    int cb = kb * 32;
    v16h A0 = load_a(Act, (size_t)(tokB*32 +      m), 1024, cb, half);
    v16h A1 = load_a(Act, (size_t)(tokB*32 + 16 + m), 1024, cb, half);
    v16h B0 = load_b(Wm,  (size_t)(outB*32 +      m), 1024, cb, half);
    v16h B1 = load_b(Wm,  (size_t)(outB*32 + 16 + m), 1024, cb, half);
    acc[0][0] = wmma_f16(A0, B0, acc[0][0]);
    acc[0][1] = wmma_f16(A0, B1, acc[0][1]);
    acc[1][0] = wmma_f16(A1, B0, acc[1][0]);
    acc[1][1] = wmma_f16(A1, B1, acc[1][1]);
  }
#pragma unroll
  for (int j = 0; j < 2; ++j) {
    int oc = outB*32 + j*16 + m;          // lane's output column
    float bz = bias[oc];
#pragma unroll
    for (int i = 0; i < 2; ++i) {
      int tb = tokB*32 + i*16 + half*8;   // 8 consecutive tokens
      float r[8];
#pragma unroll
      for (int e = 0; e < 8; ++e) r[e] = acc[i][j][e] + bz;
      uint4 st = {pkh(r[0],r[1]), pkh(r[2],r[3]), pkh(r[4],r[5]), pkh(r[6],r[7])};
      *reinterpret_cast<uint4*>(OutT + (size_t)oc*4096 + tb) = st;
    }
  }
}

// ---------------------------------------------------------------------------
// LDS staging helper: one 16B chunk, async DMA if available.
// Builtin signature (from compiler diagnostic): param0 is int4 addrspace(1)*.
__device__ __forceinline__ void stage16(const _Float16* gp, _Float16* lp) {
#if USE_ASYNC_LDS
  __builtin_amdgcn_global_load_async_to_lds_b128(
      (as1_v4i_p)(v4i*)(void*)const_cast<_Float16*>(gp),
      (as3_v4i_p)(v4i*)(void*)lp, 0, 0);
#else
  *reinterpret_cast<uint4*>(lp) = *reinterpret_cast<const uint4*>(gp);
#endif
}

// 5) Flash attention (transposed): per workgroup = (b, h, 4 q-tiles, one per wave).
//    K/V 32-key panels staged to LDS, shared by the 4 waves.
//    scores^T tile: D[m=key,n=q] = K[key,hd] . Q[q,hd]     (Q pre-scaled by SCALE*log2e)
//    ctx^T tile:    D[m=hd,n=q] += V[key,hd] * P[q,key]
__global__ __launch_bounds__(128) void k_attn(const _Float16* __restrict__ Q,
                                              const _Float16* __restrict__ K,
                                              const _Float16* __restrict__ VT,
                                              const unsigned int* __restrict__ mb,
                                              _Float16* __restrict__ Ctx) {
  __shared__ __align__(16) _Float16 Kp[32 * 72];  // [32 keys][64 hd], stride 72 halves
  __shared__ __align__(16) _Float16 Vp[64 * 40];  // [64 hd][32 keys], stride 40 halves
  const int b = blockIdx.z, h = blockIdx.y;
  const int wave = threadIdx.x >> 5, lane = threadIdx.x & 31;
  const int m = lane & 15, half = lane >> 4;
  const int qtile = blockIdx.x * 4 + wave;
  const int q = qtile * 16 + m;                    // query index within S
  const size_t qglob = (size_t)b * 2048 + q;       // token index
  // Q^T B-operands, reused across the whole key loop.
  v16h Bq0 = load_b(Q, qglob, 1024, h * 64,      half);
  v16h Bq1 = load_b(Q, qglob, 1024, h * 64 + 32, half);
  v8f accv[4] = {};
  float mi = -1e30f, li = 0.f;
  for (int kb = 0; kb < 64; ++kb) {
    __syncthreads();  // protect LDS panels from previous iteration's readers
    // stage K panel: 32 rows x 128B  (async DMA straight into LDS when available)
    for (int c = threadIdx.x; c < 256; c += 128) {
      int row = c >> 3, col = c & 7;
      stage16(K + ((size_t)b*2048 + kb*32 + row) * 1024 + h*64 + col*8,
              Kp + row*72 + col*8);
    }
    // stage V panel: 64 rows x 64B
    for (int c = threadIdx.x; c < 256; c += 128) {
      int row = c >> 2, col = c & 3;
      stage16(VT + (size_t)(h*64 + row) * 4096 + b*2048 + kb*32 + col*8,
              Vp + row*40 + col*8);
    }
#if USE_ASYNC_LDS
    __builtin_amdgcn_s_wait_asynccnt(0);  // drain this wave's async DMAs
#endif
    __syncthreads();
    if (kb + 1 < 64) {  // pull next panels toward L2
      __builtin_prefetch(K + ((size_t)b*2048 + (kb+1)*32 + (threadIdx.x & 31)) * 1024 + h*64, 0, 1);
      __builtin_prefetch(VT + (size_t)(h*64 + (threadIdx.x & 63)) * 4096 + b*2048 + (kb+1)*32, 0, 1);
    }
    // scores^T, two 16-key tiles, hd=64 -> two WMMAs each
    v8f s0 = {}, s1 = {};
    s0 = wmma_f16(load_a(Kp, (size_t)(m),      72, 0,  half), Bq0, s0);
    s0 = wmma_f16(load_a(Kp, (size_t)(m),      72, 32, half), Bq1, s0);
    s1 = wmma_f16(load_a(Kp, (size_t)(16 + m), 72, 0,  half), Bq0, s1);
    s1 = wmma_f16(load_a(Kp, (size_t)(16 + m), 72, 32, half), Bq1, s1);
    // mask (branchless selects; EXEC stays all-ones for WMMA)
    unsigned int mw = mb[(size_t)q * 64 + kb];
#pragma unroll
    for (int r = 0; r < 8; ++r) {
      s0[r] = ((mw >> (r + 8*half)) & 1u)      ? s0[r] : -1e30f;
      s1[r] = ((mw >> (16 + r + 8*half)) & 1u) ? s1[r] : -1e30f;
    }
    // online softmax (log2 domain; SCALE*log2e folded into Q)
    float bm = -1e30f;
#pragma unroll
    for (int r = 0; r < 8; ++r) bm = fmaxf(bm, fmaxf(s0[r], s1[r]));
    bm = fmaxf(bm, __shfl_xor(bm, 16, 32));
    float mn = fmaxf(mi, bm);
    float corr = exp2f(mi - mn);
    float p0[8], p1[8]; float ps = 0.f;
#pragma unroll
    for (int r = 0; r < 8; ++r) {
      p0[r] = exp2f(s0[r] - mn);
      p1[r] = exp2f(s1[r] - mn);
      ps += p0[r] + p1[r];
    }
    ps += __shfl_xor(ps, 16, 32);
    li = li * corr + ps;
    mi = mn;
    // P^T -> B operand.  B wants K(key) = 16*half + 2v; own half supplies 8 keys,
    // partner lane (xor 16) supplies the other 8 -> exchange packed pairs.
    unsigned int pk0[4], pk1[4], o0[4], o1[4];
#pragma unroll
    for (int v = 0; v < 4; ++v) {
      pk0[v] = pkh(p0[2*v], p0[2*v+1]);
      pk1[v] = pkh(p1[2*v], p1[2*v+1]);
    }
#pragma unroll
    for (int v = 0; v < 4; ++v) {
      o0[v] = (unsigned int)__shfl_xor((int)pk0[v], 16, 32);
      o1[v] = (unsigned int)__shfl_xor((int)pk1[v], 16, 32);
    }
    v8u bu;
#pragma unroll
    for (int v = 0; v < 4; ++v) {
      bu[v]     = half ? o1[v]  : pk0[v];
      bu[v + 4] = half ? pk1[v] : o0[v];
    }
    v16h Bp = __builtin_bit_cast(v16h, bu);
    // rescale running context, accumulate V^T x P^T (4 hd tiles)
#pragma unroll
    for (int t = 0; t < 4; ++t) {
#pragma unroll
      for (int e = 0; e < 8; ++e) accv[t][e] *= corr;
      accv[t] = wmma_f16(load_a(Vp, (size_t)(t*16 + m), 40, 0, half), Bp, accv[t]);
    }
  }
  // finalize: ctx[q][h*64 + hd] = acc/l  (8 consecutive hd per lane -> b128)
  float inv = 1.f / li;
#pragma unroll
  for (int t = 0; t < 4; ++t) {
    float r[8];
#pragma unroll
    for (int e = 0; e < 8; ++e) r[e] = accv[t][e] * inv;
    uint4 st = {pkh(r[0],r[1]), pkh(r[2],r[3]), pkh(r[4],r[5]), pkh(r[6],r[7])};
    *reinterpret_cast<uint4*>(Ctx + qglob*1024 + h*64 + t*16 + half*8) = st;
  }
}

// ---------------------------------------------------------------------------
extern "C" void kernel_launch(void* const* d_in, const int* in_sizes, int n_in,
                              void* d_out, int out_size, void* d_ws, size_t ws_size,
                              hipStream_t stream) {
  const float* x    = (const float*)d_in[0];
  const int*   mask = (const int*)d_in[1];
  const float* Wq   = (const float*)d_in[2];
  const float* bq   = (const float*)d_in[3];
  const float* Wk   = (const float*)d_in[4];
  const float* bk   = (const float*)d_in[5];
  const float* Wv   = (const float*)d_in[6];
  const float* bv   = (const float*)d_in[7];
  const float* Wo   = (const float*)d_in[8];
  const float* bo   = (const float*)d_in[9];

  char* ws = (char*)d_ws;
  const size_t MiB = 1ull << 20;
  _Float16* Xh = (_Float16*)(ws + 0);         // 8 MiB  [4096][1024]
  _Float16* Mq = (_Float16*)(ws + 8*MiB);     // 2 MiB  [1024][1024]
  _Float16* Mk = (_Float16*)(ws + 10*MiB);
  _Float16* Mv = (_Float16*)(ws + 12*MiB);
  _Float16* Mo = (_Float16*)(ws + 14*MiB);
  _Float16* Qh = (_Float16*)(ws + 16*MiB);    // 8 MiB  [tok][emb]
  _Float16* Kh = (_Float16*)(ws + 24*MiB);    // 8 MiB  [tok][emb]
  _Float16* Vt = (_Float16*)(ws + 32*MiB);    // 8 MiB  [emb][tok]
  _Float16* Cx = (_Float16*)(ws + 40*MiB);    // 8 MiB  [tok][emb]
  unsigned int* mbits = (unsigned int*)(ws + 48*MiB);  // 512 KiB

  k_prep_x  <<<4096, 256, 0, stream>>>(x, Xh);
  k_maskbits<<<512,  256, 0, stream>>>(mask, mbits);
  k_expand  <<<256,  256, 0, stream>>>(Wq, Mq);
  k_expand  <<<256,  256, 0, stream>>>(Wk, Mk);
  k_expand  <<<256,  256, 0, stream>>>(Wv, Mv);
  k_expand  <<<256,  256, 0, stream>>>(Wo, Mo);

  const float qscale = 0.125f * 1.44269504088896341f;  // 1/sqrt(64) * log2(e)
  k_gemm_T<false><<<1024, 128, 0, stream>>>(Mq, Xh, bq, qscale, (void*)Qh);
  k_gemm_T<false><<<1024, 128, 0, stream>>>(Mk, Xh, bk, 1.0f,   (void*)Kh);
  k_gemm_N       <<<1024, 128, 0, stream>>>(Xh, Mv, bv, Vt);

  k_attn<<<dim3(32, 16, 2), 128, 0, stream>>>(Qh, Kh, Vt, mbits, Cx);

  k_gemm_T<true><<<1024, 128, 0, stream>>>(Mo, Cx, bo, 1.0f, d_out);
}
